// BinaryConnectNet_52158082843177
// MI455X (gfx1250) — matrix operand order repaired and checked by reading, not compiled, not run on hardware
//
#include <hip/hip_runtime.h>
#include <math.h>

typedef __attribute__((ext_vector_type(16))) _Float16 v16h;
typedef __attribute__((ext_vector_type(8)))  float    v8f;

#define EPSV 1e-5f
#define NB   512          // batch
#define HW   256          // 16*16 pixels per feature map
#define NPIX (NB * HW)    // 131072 GEMM columns
#define FCK  65536        // FC reduction length (256*16*16)

__device__ __forceinline__ float bsign(float w) { return w >= 0.0f ? 1.0f : -1.0f; }
__device__ __forceinline__ float bnrelu(float v, float sc, float bi) {
    float r = fmaf(v, sc, bi);
    return r > 0.0f ? r : 0.0f;
}

// ---------------------------------------------------------------------------
// Block 1: binary conv 3x3 (3->64, pad 1) on 32x32, fused 2x2 maxpool + clip.
// ---------------------------------------------------------------------------
__global__ void conv1_pool_clip(const float* __restrict__ X,   // [512,3,32,32]
                                const float* __restrict__ W,   // [64,3,3,3]
                                float* __restrict__ Y) {       // [512,64,16,16]
    long i = (long)blockIdx.x * blockDim.x + threadIdx.x;
    if (i >= (long)NB * 64 * HW) return;
    int p  = (int)(i & 255);
    int c  = (int)((i >> 8) & 63);
    int n  = (int)(i >> 14);
    int py = p >> 4, px = p & 15;
    float best = -INFINITY;
    for (int sy = 0; sy < 2; ++sy) {
        for (int sx = 0; sx < 2; ++sx) {
            int oy = py * 2 + sy, ox = px * 2 + sx;
            float acc = 0.0f;
            for (int ci = 0; ci < 3; ++ci) {
                const float* xin = X + ((long)n * 3 + ci) * 1024;
                const float* wk  = W + ((long)c * 3 + ci) * 9;
                #pragma unroll
                for (int ky = 0; ky < 3; ++ky) {
                    int yy = oy + ky - 1;
                    if (yy < 0 || yy > 31) continue;
                    #pragma unroll
                    for (int kx = 0; kx < 3; ++kx) {
                        int xx = ox + kx - 1;
                        if (xx < 0 || xx > 31) continue;
                        acc += bsign(wk[ky * 3 + kx]) * xin[yy * 32 + xx];
                    }
                }
            }
            best = fmaxf(best, acc);
        }
    }
    Y[i] = fminf(fmaxf(best, -128.0f), 127.0f);
}

// ---------------------------------------------------------------------------
// Shift-BN statistics: per-channel mean / population variance over N*H*W.
// ---------------------------------------------------------------------------
__global__ void bn_stats(const float* __restrict__ X, int C,
                         float* __restrict__ mean, float* __restrict__ var) {
    const int c = blockIdx.x;
    const int NP = NB * HW;
    float s = 0.0f, s2 = 0.0f;
    for (int i = threadIdx.x; i < NP; i += blockDim.x) {
        int n = i >> 8, p = i & 255;
        float v = X[((long)n * C + c) * HW + p];
        s += v; s2 += v * v;
    }
    __shared__ float sh[256], sh2[256];
    sh[threadIdx.x] = s; sh2[threadIdx.x] = s2;
    __syncthreads();
    for (int off = 128; off > 0; off >>= 1) {
        if ((int)threadIdx.x < off) {
            sh[threadIdx.x]  += sh[threadIdx.x + off];
            sh2[threadIdx.x] += sh2[threadIdx.x + off];
        }
        __syncthreads();
    }
    if (threadIdx.x == 0) {
        float m = sh[0] / (float)NP;
        mean[c] = m;
        var[c]  = sh2[0] / (float)NP - m * m;
    }
}

// Fold AP2(1/std) shift + gamma/beta into per-channel scale/bias.
__global__ void bn_finalize(const float* __restrict__ mean, const float* __restrict__ var,
                            const float* __restrict__ g, const float* __restrict__ b,
                            int C, float* __restrict__ scale, float* __restrict__ bias) {
    int c = blockIdx.x * blockDim.x + threadIdx.x;
    if (c >= C) return;
    float shift = exp2f(roundf(log2f(rsqrtf(var[c] + EPSV))));
    float s = shift * g[c];
    scale[c] = s;
    bias[c]  = b[c] - mean[c] * s;
}

// ---------------------------------------------------------------------------
// Fused BN+ReLU + residual depthwise binary 3x3, output packed as f16 in
// WMMA B-fragment tile order so the GEMM can async-copy it verbatim to LDS:
//   tile ct = col/16 holds Cin*16 halves:  [(k/32)*512 + lane*16 + (k%16)],
//   lane = (col%16) + 16*((k>>4)&1).
// ---------------------------------------------------------------------------
__global__ void dw_residual_bn_pack(const float* __restrict__ X, const float* __restrict__ W,
                                    int C, const float* __restrict__ scale,
                                    const float* __restrict__ bias,
                                    _Float16* __restrict__ Yf) {
    long i = (long)blockIdx.x * blockDim.x + threadIdx.x;
    long total = (long)NB * C * HW;
    if (i >= total) return;
    int p = (int)(i & 255);
    int c = (int)((i >> 8) % C);
    int n = (int)(i / ((long)C * HW));
    int y = p >> 4, x = p & 15;
    const float* base = X + (i - p);
    const float* wk   = W + (long)c * 9;
    const float sc = scale[c], bi = bias[c];
    float acc = bnrelu(base[p], sc, bi);
    #pragma unroll
    for (int ky = 0; ky < 3; ++ky) {
        int yy = y + ky - 1;
        if (yy < 0 || yy > 15) continue;
        #pragma unroll
        for (int kx = 0; kx < 3; ++kx) {
            int xx = x + kx - 1;
            if (xx < 0 || xx > 15) continue;
            acc += bsign(wk[ky * 3 + kx]) * bnrelu(base[yy * 16 + xx], sc, bi);
        }
    }
    long col = (long)n * HW + p;
    long ct  = col >> 4;
    int  cl  = (int)(col & 15);
    int  lane = cl + 16 * ((c >> 4) & 1);
    long off = ct * ((long)C * 16) + ((c >> 5) * 512 + lane * 16 + (c & 15));
    Yf[off] = (_Float16)acc;
}

// ---------------------------------------------------------------------------
// Weight prep: binarize + pack into WMMA A-fragment order (f16).
// ---------------------------------------------------------------------------
__global__ void prep_wfrag(const float* __restrict__ W, int Cin, int Cout,
                           _Float16* __restrict__ WF) {
    int i = blockIdx.x * blockDim.x + threadIdx.x;
    if (i >= Cout * Cin) return;
    int e    = i & 15;
    int lane = (i >> 4) & 31;
    int tile = i >> 9;
    int KT = Cin >> 5;
    int mt = tile / KT, kt = tile - mt * KT;
    int half = lane >> 4, lr = lane & 15;
    int m = mt * 16 + lr;
    int k = kt * 32 + half * 16 + e;
    WF[i] = (_Float16)bsign(W[(long)m * Cin + k]);
}

// FC weights into B-fragment order (f16 +-1, zero padding for cols 10..15).
__global__ void prep_wfc(const float* __restrict__ W, _Float16* __restrict__ WF) {
    int i = blockIdx.x * blockDim.x + threadIdx.x;
    if (i >= (FCK / 32) * 512) return;
    int e    = i & 15;
    int lane = (i >> 4) & 31;
    int tile = i >> 9;
    int half = lane >> 4, col = lane & 15;
    int k = tile * 32 + half * 16 + e;
    WF[i] = (col < 10) ? (_Float16)bsign(W[(long)col * FCK + k]) : (_Float16)0.0f;
}

// ---------------------------------------------------------------------------
// 1x1 binary conv as WMMA GEMM. The B tile (Cin x 16 cols, f16, fragment
// order) is a contiguous global region -> copied to LDS with
// GLOBAL_LOAD_ASYNC_TO_LDS_B128 (ASYNCcnt), then shared by all M-tiles.
// ---------------------------------------------------------------------------
template<int Cin, int Cout>
__global__ __launch_bounds__(256) void gemm1x1_wmma(
        const _Float16* __restrict__ Xf,      // packed B tiles [NPIX/16][Cin*16]
        const _Float16* __restrict__ WF,      // A fragments
        float* __restrict__ Y) {              // [512,Cout,16,16]
    constexpr int KT = Cin / 32;
    constexpr int MT = Cout / 16;
    __shared__ _Float16 Bs[Cin * 16];

    const int t = threadIdx.x;                // 0..255
    // Async copy: Cin*32 bytes = Cin*2 x 16B chunks, one b128 per lane issue.
    {
        unsigned ldsBase = (unsigned)(size_t)(void*)&Bs[0];
        const char* g0 = (const char*)(Xf + (long)blockIdx.x * (Cin * 16));
        for (int idx = t; idx < Cin * 2; idx += 256) {
            unsigned          l = ldsBase + idx * 16;
            unsigned long long g = (unsigned long long)(g0 + idx * 16);
            asm volatile("global_load_async_to_lds_b128 %0, %1, off"
                         :: "v"(l), "v"(g) : "memory");
        }
        asm volatile("s_wait_asynccnt 0x0" ::: "memory");
    }
    __syncthreads();

    const int wave = t >> 5;
    const int lane = t & 31;
    const int half = lane >> 4, lr = lane & 15;
    const int n0   = blockIdx.x * 16;
    const int nImg = n0 >> 8;
    const int p0   = n0 & 255;

    for (int mt = wave; mt < MT; mt += 8) {
        v8f acc = {};
        #pragma unroll
        for (int kt = 0; kt < KT; ++kt) {
            v16h a = *(const v16h*)&WF[((long)(mt * KT + kt) * 32 + lane) * 16];
            v16h b = *(const v16h*)&Bs[(kt * 32 + lane) * 16];
            acc = __builtin_amdgcn_wmma_f32_16x16x32_f16(false, a, false, b,
                                                         (short)0, acc, false, false);
        }
        #pragma unroll
        for (int r = 0; r < 8; ++r) {
            int m = mt * 16 + half * 8 + r;
            float v = fminf(fmaxf(acc[r], -128.0f), 127.0f);
            Y[((long)nImg * Cout + m) * HW + (p0 + lr)] = v;
        }
    }
}

// ---------------------------------------------------------------------------
// FC head with BN4 fused into the A load; streaming A prefetched.
// ---------------------------------------------------------------------------
__global__ void fc_bias_init(float* __restrict__ Out, const float* __restrict__ bfc) {
    int i = blockIdx.x * blockDim.x + threadIdx.x;
    if (i < NB * 10) Out[i] = bfc[i % 10];
}

__global__ void fc_wmma(const float* __restrict__ A,       // [512, 65536]
                        const _Float16* __restrict__ WFC,  // B fragments
                        const float* __restrict__ scale,   // BN4 scale [256]
                        const float* __restrict__ bias,    // BN4 bias  [256]
                        float* __restrict__ Out) {         // [512, 10]
    const int lane = threadIdx.x;
    const int half = lane >> 4;
    const int lr   = lane & 15;
    const int m0   = blockIdx.x * 16;
    const long kbase = (long)blockIdx.y * 4096;
    const float* Arow = A + (long)(m0 + lr) * FCK;

    v8f acc = {};
    for (int k0 = 0; k0 < 4096; k0 += 32) {
        const long kb = kbase + k0;
        __builtin_prefetch(Arow + kb + 256, 0, 1);   // global_prefetch_b8
        v16h a;
        #pragma unroll
        for (int h = 0; h < 2; ++h) {
            long s = kb + h * 8 + half * 16;    // 8 consecutive K, one channel
            int  c = (int)(s >> 8);
            float sc = scale[c], bi = bias[c];
            float4 u = *(const float4*)(Arow + s);
            float4 v = *(const float4*)(Arow + s + 4);
            a[h * 8 + 0] = (_Float16)bnrelu(u.x, sc, bi);
            a[h * 8 + 1] = (_Float16)bnrelu(u.y, sc, bi);
            a[h * 8 + 2] = (_Float16)bnrelu(u.z, sc, bi);
            a[h * 8 + 3] = (_Float16)bnrelu(u.w, sc, bi);
            a[h * 8 + 4] = (_Float16)bnrelu(v.x, sc, bi);
            a[h * 8 + 5] = (_Float16)bnrelu(v.y, sc, bi);
            a[h * 8 + 6] = (_Float16)bnrelu(v.z, sc, bi);
            a[h * 8 + 7] = (_Float16)bnrelu(v.w, sc, bi);
        }
        v16h b = *(const v16h*)&WFC[((kb >> 5) * 32 + lane) * 16];
        acc = __builtin_amdgcn_wmma_f32_16x16x32_f16(false, a, false, b,
                                                     (short)0, acc, false, false);
    }
    if (lr < 10) {
        #pragma unroll
        for (int r = 0; r < 8; ++r) {
            int m = m0 + half * 8 + r;
            atomicAdd(&Out[m * 10 + lr], acc[r]);
        }
    }
}

// ---------------------------------------------------------------------------
// Launch
// ---------------------------------------------------------------------------
extern "C" void kernel_launch(void* const* d_in, const int* in_sizes, int n_in,
                              void* d_out, int out_size, void* d_ws, size_t ws_size,
                              hipStream_t stream) {
    const float* x   = (const float*)d_in[0];
    const float* w1  = (const float*)d_in[1];
    const float* w21 = (const float*)d_in[2];
    const float* w22 = (const float*)d_in[3];
    const float* w31 = (const float*)d_in[4];
    const float* w32 = (const float*)d_in[5];
    const float* w41 = (const float*)d_in[6];
    const float* w42 = (const float*)d_in[7];
    const float* g1  = (const float*)d_in[8];
    const float* b1  = (const float*)d_in[9];
    const float* g2  = (const float*)d_in[10];
    const float* b2  = (const float*)d_in[11];
    const float* g3  = (const float*)d_in[12];
    const float* b3  = (const float*)d_in[13];
    const float* g4  = (const float*)d_in[14];
    const float* b4  = (const float*)d_in[15];
    const float* wfc = (const float*)d_in[16];
    const float* bfc = (const float*)d_in[17];
    float* out = (float*)d_out;

    const long BIG = (long)NB * 256 * HW;           // 33,554,432 elements
    float*    bufA = (float*)d_ws;                  // f32 conv outputs (134 MB)
    _Float16* bufF = (_Float16*)(bufA + BIG);       // packed f16 GEMM input (67 MB)
    float*    mean = (float*)(bufF + BIG);
    float*    var   = mean + 256;
    float*    scale = var + 256;
    float*    bias  = scale + 256;
    _Float16* wf22 = (_Float16*)(bias + 256);       // 4096 halves
    _Float16* wf32 = wf22 + 64 * 64;                // 16384 halves
    _Float16* wf42 = wf32 + 256 * 64;               // 65536 halves
    _Float16* wffc = wf42 + 256 * 256;              // 1M halves

    auto blocks = [](long n, int t) { return (unsigned)((n + t - 1) / t); };

    // Weight prep (small, L2-resident afterwards)
    prep_wfrag<<<blocks(64 * 64, 256), 256, 0, stream>>>(w22, 64, 64, wf22);
    prep_wfrag<<<blocks(256 * 64, 256), 256, 0, stream>>>(w32, 64, 256, wf32);
    prep_wfrag<<<blocks(256 * 256, 256), 256, 0, stream>>>(w42, 256, 256, wf42);
    prep_wfc<<<blocks((FCK / 32) * 512, 256), 256, 0, stream>>>(wfc, wffc);

    // ---- block 1 ----
    conv1_pool_clip<<<blocks((long)NB * 64 * HW, 256), 256, 0, stream>>>(x, w1, bufA);
    bn_stats<<<64, 256, 0, stream>>>(bufA, 64, mean, var);
    bn_finalize<<<1, 64, 0, stream>>>(mean, var, g1, b1, 64, scale, bias);

    // ---- block 2: 64 -> 64 ----
    dw_residual_bn_pack<<<blocks((long)NB * 64 * HW, 256), 256, 0, stream>>>(bufA, w21, 64, scale, bias, bufF);
    gemm1x1_wmma<64, 64><<<NPIX / 16, 256, 0, stream>>>(bufF, wf22, bufA);
    bn_stats<<<64, 256, 0, stream>>>(bufA, 64, mean, var);
    bn_finalize<<<1, 64, 0, stream>>>(mean, var, g2, b2, 64, scale, bias);

    // ---- block 3: 64 -> 256 ----
    dw_residual_bn_pack<<<blocks((long)NB * 64 * HW, 256), 256, 0, stream>>>(bufA, w31, 64, scale, bias, bufF);
    gemm1x1_wmma<64, 256><<<NPIX / 16, 256, 0, stream>>>(bufF, wf32, bufA);
    bn_stats<<<256, 256, 0, stream>>>(bufA, 256, mean, var);
    bn_finalize<<<1, 256, 0, stream>>>(mean, var, g3, b3, 256, scale, bias);

    // ---- block 4: 256 -> 256 ----
    dw_residual_bn_pack<<<blocks((long)NB * 256 * HW, 256), 256, 0, stream>>>(bufA, w41, 256, scale, bias, bufF);
    gemm1x1_wmma<256, 256><<<NPIX / 16, 256, 0, stream>>>(bufF, wf42, bufA);
    bn_stats<<<256, 256, 0, stream>>>(bufA, 256, mean, var);
    bn_finalize<<<1, 256, 0, stream>>>(mean, var, g4, b4, 256, scale, bias);

    // ---- FC head ----
    fc_bias_init<<<blocks(NB * 10, 256), 256, 0, stream>>>(out, bfc);
    fc_wmma<<<dim3(NB / 16, 16), 32, 0, stream>>>(bufA, wffc, scale, bias, out);
}
